// LSTM_61899068670344
// MI455X (gfx1250) — matrix operand order repaired and checked by reading, not compiled, not run on hardware
//
#include <hip/hip_runtime.h>

#define Bb 32
#define Tt 512
#define Ee 300
#define EP 320
#define Hh 1024
#define G4 4096
#define BT (Bb*Tt)

typedef __attribute__((ext_vector_type(16))) __bf16 v16bf;
typedef __attribute__((ext_vector_type(8)))  float v8f;

union Frag16 { v16bf v; uint4 q[2]; };

__device__ __forceinline__ unsigned short f2bf(float f) {
  unsigned u = __float_as_uint(f);
  u += 0x7FFFu + ((u >> 16) & 1u);   // round-to-nearest-even
  return (unsigned short)(u >> 16);
}
__device__ __forceinline__ float sigf(float x) { return 1.0f / (1.0f + __expf(-x)); }

// ---------------- utility kernels ----------------
__global__ void k_zero(unsigned* p, size_t nwords) {
  for (size_t i = (size_t)blockIdx.x * blockDim.x + threadIdx.x; i < nwords;
       i += (size_t)gridDim.x * blockDim.x) p[i] = 0u;
}

__global__ void k_cvt(const float* __restrict__ src, unsigned short* __restrict__ dst,
                      int rows, int K, int Kpad) {
  size_t total = (size_t)rows * Kpad;
  for (size_t i = (size_t)blockIdx.x * blockDim.x + threadIdx.x; i < total;
       i += (size_t)gridDim.x * blockDim.x) {
    int r = (int)(i / Kpad), k = (int)(i % Kpad);
    dst[i] = (k < K) ? f2bf(src[(size_t)r * K + k]) : (unsigned short)0;
  }
}

__global__ void k_embed(const int* __restrict__ tok, const float* __restrict__ emb,
                        unsigned short* __restrict__ x0) {
  int bt = blockIdx.x;
  int e  = threadIdx.x;          // 320 threads
  int tk = tok[bt];
  unsigned short v = 0;
  if (e < Ee) v = f2bf(emb[(size_t)tk * Ee + e]);
  x0[(size_t)bt * EP + e] = v;
}

// ---------------- input-projection GEMM: Y[M,4096] = Xbf16[M,Kpad] @ W^T + b_ih + b_hh ----
// WG tile 64(M) x 256(N); 8 waves as 2x4; wave tile 32x64 -> 8 accumulators,
// 12 b128 loads : 8 WMMA per k-chunk.
__global__ __launch_bounds__(256) void k_gemm_xp(const unsigned short* __restrict__ X,
                                                 const unsigned short* __restrict__ W,
                                                 const float* __restrict__ b0,
                                                 const float* __restrict__ b1,
                                                 float* __restrict__ Y, int Kpad) {
  const int lane = threadIdx.x & 31;
  const int wave = threadIdx.x >> 5;
  const int wm = wave >> 2, wn = wave & 3;           // 2 x 4 waves
  const int mbase = blockIdx.y * 64 + wm * 32;
  const int nbase = blockIdx.x * 256 + wn * 64;
  const int lo = lane & 15, hi = lane >> 4;

  v8f acc[2][4] = {};
  const int nk = Kpad >> 5;
  for (int kc = 0; kc < nk; ++kc) {
    const int kbA = (kc << 5) + (hi << 4);
    const int kbB = (kc << 5) + (hi << 4);
    Frag16 a[2], bm[4];
#pragma unroll
    for (int mt = 0; mt < 2; ++mt) {
      const uint4* p = (const uint4*)(X + (size_t)(mbase + mt * 16 + lo) * Kpad + kbA);
      a[mt].q[0] = p[0]; a[mt].q[1] = p[1];          // 16 contiguous K per lane-half
    }
#pragma unroll
    for (int nt = 0; nt < 4; ++nt) {
      const uint4* p = (const uint4*)(W + (size_t)(nbase + nt * 16 + lo) * Kpad + kbB);
      bm[nt].q[0] = p[0]; bm[nt].q[1] = p[1];        // 16 contiguous K per lane-half
    }
#pragma unroll
    for (int mt = 0; mt < 2; ++mt)
#pragma unroll
      for (int nt = 0; nt < 4; ++nt)
        acc[mt][nt] = __builtin_amdgcn_wmma_f32_16x16x32_bf16(
            false, a[mt].v, false, bm[nt].v, (short)0, acc[mt][nt], false, false);
  }
#pragma unroll
  for (int nt = 0; nt < 4; ++nt) {
    const int n = nbase + nt * 16 + lo;
    const float bias = b0[n] + b1[n];
#pragma unroll
    for (int mt = 0; mt < 2; ++mt)
#pragma unroll
      for (int r = 0; r < 8; ++r) {
        const int m = mbase + mt * 16 + r + (hi << 3);
        Y[(size_t)m * G4 + n] = acc[mt][nt][r] + bias;
      }
  }
}

// ---------------- device-scope barrier over gridDim.x workgroups of one direction -------
__device__ void gridbar(unsigned* cnt, unsigned* gen, unsigned nwg) {
  __syncthreads();
  if (threadIdx.x == 0) {
    __threadfence();
    unsigned g = __hip_atomic_load(gen, __ATOMIC_RELAXED, __HIP_MEMORY_SCOPE_AGENT);
    unsigned prev = __hip_atomic_fetch_add(cnt, 1u, __ATOMIC_ACQ_REL, __HIP_MEMORY_SCOPE_AGENT);
    if (prev == nwg - 1) {
      __hip_atomic_store(cnt, 0u, __ATOMIC_RELAXED, __HIP_MEMORY_SCOPE_AGENT);
      __threadfence();
      __hip_atomic_fetch_add(gen, 1u, __ATOMIC_ACQ_REL, __HIP_MEMORY_SCOPE_AGENT);
    } else {
      while (__hip_atomic_load(gen, __ATOMIC_ACQUIRE, __HIP_MEMORY_SCOPE_AGENT) == g)
        __builtin_amdgcn_s_sleep(1);
    }
  }
  __syncthreads();
}

// ---------------- persistent bidirectional LSTM recurrence ------------------------------
// grid = (8, 2): 8 WGs per direction (each owns 128 h-columns), blockIdx.y = direction.
// Each of the 8 waves owns 16 h-columns -> computes its own i/f/g/o gate tiles and keeps
// h and c in accumulator-layout registers for all 512 steps. xp gate loads for the step
// are issued BEFORE the K-loop so their L2 latency is hidden under the WMMA pipeline.
__global__ __launch_bounds__(256) void k_lstm(const float* __restrict__ xp_f,
                                              const float* __restrict__ xp_b,
                                              const unsigned short* __restrict__ whh_f,
                                              const unsigned short* __restrict__ whh_b,
                                              const int* __restrict__ lengths,
                                              unsigned short* __restrict__ hbuf,  // [dir][2][32*1024] bf16
                                              float* __restrict__ hfinal,         // [dir][32*1024] f32
                                              unsigned short* __restrict__ seq,   // x1 (BTx2048) or null
                                              unsigned* __restrict__ bar) {
  const int dir = blockIdx.y;
  const float* xp = dir ? xp_b : xp_f;
  const unsigned short* W = dir ? whh_b : whh_f;
  unsigned short* h0 = hbuf + (size_t)dir * 2 * Bb * Hh;
  unsigned short* h1 = h0 + Bb * Hh;
  unsigned* cnt = bar + dir * 2;
  unsigned* gen = cnt + 1;

  __shared__ int s_len[Bb];
  if (threadIdx.x < Bb) s_len[threadIdx.x] = lengths[threadIdx.x];
  __syncthreads();

  const int lane = threadIdx.x & 31;
  const int wave = threadIdx.x >> 5;
  const int lo = lane & 15, hi = lane >> 4;
  const int j16 = blockIdx.x * 128 + wave * 16;   // this wave's 16 h-columns
  const int n = j16 + lo;

  v8f c[2] = {}, h[2] = {};

  for (int s = 0; s < Tt; ++s) {
    unsigned short* hc = (s & 1) ? h1 : h0;
    unsigned short* hn = (s & 1) ? h0 : h1;

    // ---- issue this step's xp gate loads up front (latency hidden by GEMM below) ----
    float xi[2][8], xf[2][8], xg[2][8], xo[2][8];
#pragma unroll
    for (int mt = 0; mt < 2; ++mt)
#pragma unroll
      for (int r = 0; r < 8; ++r) {
        const int m = mt * 16 + r + (hi << 3);
        const int len = s_len[m];
        int t = dir ? (len - 1 - s) : s;
        if (t < 0) t = 0;
        const float* xr = xp + ((size_t)m * Tt + t) * G4;
        xi[mt][r] = xr[n];
        xf[mt][r] = xr[Hh + n];
        xg[mt][r] = xr[2 * Hh + n];
        xo[mt][r] = xr[3 * Hh + n];
      }

    v8f acc[2][4];
#pragma unroll
    for (int mt = 0; mt < 2; ++mt)
#pragma unroll
      for (int g = 0; g < 4; ++g) acc[mt][g] = (v8f){};

    for (int kc = 0; kc < 32; ++kc) {
      const int kbA = (kc << 5) + (hi << 4);
      const int kbB = (kc << 5) + (hi << 4);
      Frag16 a[2];
#pragma unroll
      for (int mt = 0; mt < 2; ++mt) {
        const uint4* p = (const uint4*)(hc + (size_t)(mt * 16 + lo) * Hh + kbA);
        a[mt].q[0] = p[0]; a[mt].q[1] = p[1];
      }
#pragma unroll
      for (int g = 0; g < 4; ++g) {
        Frag16 bm;
        const uint4* p = (const uint4*)(W + (size_t)(g * Hh + j16 + lo) * Hh + kbB);
        bm.q[0] = p[0]; bm.q[1] = p[1];
#pragma unroll
        for (int mt = 0; mt < 2; ++mt)
          acc[mt][g] = __builtin_amdgcn_wmma_f32_16x16x32_bf16(
              false, a[mt].v, false, bm.v, (short)0, acc[mt][g], false, false);
      }
    }

#pragma unroll
    for (int mt = 0; mt < 2; ++mt) {
#pragma unroll
      for (int r = 0; r < 8; ++r) {
        const int m = mt * 16 + r + (hi << 3);        // batch row
        const int len = s_len[m];
        const bool valid = s < len;
        const float gi = acc[mt][0][r] + xi[mt][r];
        const float gf = acc[mt][1][r] + xf[mt][r];
        const float gg = acc[mt][2][r] + xg[mt][r];
        const float go = acc[mt][3][r] + xo[mt][r];
        const float cn = sigf(gf) * c[mt][r] + sigf(gi) * tanhf(gg);
        const float hv = sigf(go) * tanhf(cn);
        if (valid) { c[mt][r] = cn; h[mt][r] = hv; }
        hn[(size_t)m * Hh + n] = f2bf(h[mt][r]);
        if (seq != nullptr && valid) {
          const int tout = dir ? (len - 1 - s) : s;
          seq[((size_t)m * Tt + tout) * (2 * Hh) + dir * Hh + n] = f2bf(hv);
        }
      }
    }
    gridbar(cnt, gen, gridDim.x);
  }

#pragma unroll
  for (int mt = 0; mt < 2; ++mt)
#pragma unroll
    for (int r = 0; r < 8; ++r) {
      const int m = mt * 16 + r + (hi << 3);
      hfinal[(size_t)dir * Bb * Hh + (size_t)m * Hh + n] = h[mt][r];
    }
}

// ---------------- final linear: out[b,c] = [h2b|h2f][b,:] . lin_w[c,:] + lin_b[c] -------
__global__ void k_final(const float* __restrict__ hfinal, const float* __restrict__ lin_w,
                        const float* __restrict__ lin_b, float* __restrict__ out) {
  const int tid = threadIdx.x;               // 64 threads
  const int b = tid >> 1, cc = tid & 1;
  const float* hf = hfinal;                  // dir 0 (forward)
  const float* hb = hfinal + Bb * Hh;        // dir 1 (backward)
  float s = lin_b[cc];
  for (int k = 0; k < Hh; ++k) s += hb[(size_t)b * Hh + k] * lin_w[(size_t)cc * 2 * Hh + k];
  for (int k = 0; k < Hh; ++k) s += hf[(size_t)b * Hh + k] * lin_w[(size_t)cc * 2 * Hh + Hh + k];
  out[b * 2 + cc] = s;
}

// ---------------- host launcher ---------------------------------------------------------
extern "C" void kernel_launch(void* const* d_in, const int* in_sizes, int n_in,
                              void* d_out, int out_size, void* d_ws, size_t ws_size,
                              hipStream_t stream) {
  (void)in_sizes; (void)n_in; (void)out_size; (void)ws_size;
  const int*   tok = (const int*)d_in[0];
  const int*   len = (const int*)d_in[1];
  const float* emb = (const float*)d_in[2];
  const float* wih0f = (const float*)d_in[3],  *whh0f = (const float*)d_in[4];
  const float* bih0f = (const float*)d_in[5],  *bhh0f = (const float*)d_in[6];
  const float* wih0b = (const float*)d_in[7],  *whh0b = (const float*)d_in[8];
  const float* bih0b = (const float*)d_in[9],  *bhh0b = (const float*)d_in[10];
  const float* wih1f = (const float*)d_in[11], *whh1f = (const float*)d_in[12];
  const float* bih1f = (const float*)d_in[13], *bhh1f = (const float*)d_in[14];
  const float* wih1b = (const float*)d_in[15], *whh1b = (const float*)d_in[16];
  const float* bih1b = (const float*)d_in[17], *bhh1b = (const float*)d_in[18];
  const float* lin_w = (const float*)d_in[19], *lin_b = (const float*)d_in[20];

  char* ws = (char*)d_ws;
  size_t off = 0;
  auto alloc = [&](size_t bytes) { size_t o = off; off = (off + bytes + 255) & ~(size_t)255; return o; };

  // zero-init state region (barrier + h ping-pong + h finals), kept contiguous
  const size_t o_bar    = alloc(256);
  const size_t o_hbuf   = alloc((size_t)2 * 2 * Bb * Hh * 2);   // bf16
  const size_t o_hfin   = alloc((size_t)2 * Bb * Hh * 4);       // f32
  const size_t zero_end = o_hfin + (size_t)2 * Bb * Hh * 4;

  const size_t o_x0    = alloc((size_t)BT * EP * 2);
  const size_t o_x1    = alloc((size_t)BT * 2 * Hh * 2);
  const size_t o_wih0f = alloc((size_t)G4 * EP * 2);
  const size_t o_wih0b = alloc((size_t)G4 * EP * 2);
  const size_t o_whh0f = alloc((size_t)G4 * Hh * 2);
  const size_t o_whh0b = alloc((size_t)G4 * Hh * 2);
  const size_t o_whh1f = alloc((size_t)G4 * Hh * 2);
  const size_t o_whh1b = alloc((size_t)G4 * Hh * 2);
  const size_t o_wih1f = alloc((size_t)G4 * 2 * Hh * 2);
  const size_t o_wih1b = alloc((size_t)G4 * 2 * Hh * 2);
  const size_t o_xpf   = alloc((size_t)BT * G4 * 4);
  const size_t o_xpb   = alloc((size_t)BT * G4 * 4);

  unsigned short* x0  = (unsigned short*)(ws + o_x0);
  unsigned short* x1  = (unsigned short*)(ws + o_x1);
  unsigned short* W0f = (unsigned short*)(ws + o_wih0f);
  unsigned short* W0b = (unsigned short*)(ws + o_wih0b);
  unsigned short* U0f = (unsigned short*)(ws + o_whh0f);
  unsigned short* U0b = (unsigned short*)(ws + o_whh0b);
  unsigned short* U1f = (unsigned short*)(ws + o_whh1f);
  unsigned short* U1b = (unsigned short*)(ws + o_whh1b);
  unsigned short* W1f = (unsigned short*)(ws + o_wih1f);
  unsigned short* W1b = (unsigned short*)(ws + o_wih1b);
  float* xpf = (float*)(ws + o_xpf);
  float* xpb = (float*)(ws + o_xpb);
  unsigned short* hbuf = (unsigned short*)(ws + o_hbuf);
  float* hfin = (float*)(ws + o_hfin);
  unsigned* bar = (unsigned*)(ws + o_bar);

  // 1) weight conversions to bf16 (padded K where needed)
  k_cvt<<<2048, 256, 0, stream>>>(wih0f, W0f, G4, Ee, EP);
  k_cvt<<<2048, 256, 0, stream>>>(wih0b, W0b, G4, Ee, EP);
  k_cvt<<<2048, 256, 0, stream>>>(whh0f, U0f, G4, Hh, Hh);
  k_cvt<<<2048, 256, 0, stream>>>(whh0b, U0b, G4, Hh, Hh);
  k_cvt<<<2048, 256, 0, stream>>>(whh1f, U1f, G4, Hh, Hh);
  k_cvt<<<2048, 256, 0, stream>>>(whh1b, U1b, G4, Hh, Hh);
  k_cvt<<<4096, 256, 0, stream>>>(wih1f, W1f, G4, 2 * Hh, 2 * Hh);
  k_cvt<<<4096, 256, 0, stream>>>(wih1b, W1b, G4, 2 * Hh, 2 * Hh);

  // 2) embedding gather -> bf16 padded
  k_embed<<<BT, EP, 0, stream>>>(tok, emb, x0);

  // 3) layer-0 input projections
  {
    dim3 g(G4 / 256, BT / 64);
    k_gemm_xp<<<g, 256, 0, stream>>>(x0, W0f, bih0f, bhh0f, xpf, EP);
    k_gemm_xp<<<g, 256, 0, stream>>>(x0, W0b, bih0b, bhh0b, xpb, EP);
  }

  // 4) layer-0 recurrence (writes sequence into x1)
  k_zero<<<256, 256, 0, stream>>>((unsigned*)(ws + o_bar), (zero_end - o_bar) / 4);
  k_lstm<<<dim3(8, 2), 256, 0, stream>>>(xpf, xpb, U0f, U0b, len, hbuf, hfin, x1, bar);

  // 5) layer-1 input projections
  {
    dim3 g(G4 / 256, BT / 64);
    k_gemm_xp<<<g, 256, 0, stream>>>(x1, W1f, bih1f, bhh1f, xpf, 2 * Hh);
    k_gemm_xp<<<g, 256, 0, stream>>>(x1, W1b, bih1b, bhh1b, xpb, 2 * Hh);
  }

  // 6) layer-1 recurrence (final hidden states only)
  k_zero<<<256, 256, 0, stream>>>((unsigned*)(ws + o_bar), (zero_end - o_bar) / 4);
  k_lstm<<<dim3(8, 2), 256, 0, stream>>>(xpf, xpb, U1f, U1b, len, hbuf, hfin, nullptr, bar);

  // 7) final linear
  k_final<<<1, 64, 0, stream>>>(hfin, lin_w, lin_b, (float*)d_out);
}
